// PotentialPredictor_68247030334587
// MI455X (gfx1250) — matrix operand order repaired and checked by reading, not compiled, not run on hardware
//
#include <hip/hip_runtime.h>
#include <hip/hip_bf16.h>

// ---------------------------------------------------------------------------
// GATv2 (4 layers, H=8, C=256) forward on gfx1250 using bf16 WMMA for all
// GEMMs and fused gather/softmax/scatter kernels for the edge phase.
// ---------------------------------------------------------------------------

#define FEAT     739
#define KPAD     768     // FEAT padded to multiple of 32
#define CCH      256     // channels
#define NHEAD    8
#define NLAYER   4
#define NGRAPH   16
#define HC       2048    // H*C
#define NEG_SLOPE 0.2f

typedef __attribute__((ext_vector_type(16))) __bf16 v16bf;
typedef __attribute__((ext_vector_type(8)))  float  v8f;

union FragBF { v16bf v; uint4 q[2]; };

// ---------------- WMMA fragment loaders (16-bit A 16x32, B 32x16) ----------

__device__ __forceinline__ v16bf load_frag_a(const __bf16* __restrict__ A, int lda,
                                             int m0, int k0, int lane) {
  // lanes 0-15: rows M=lane, K = k0+0..7 and k0+16..23
  // lanes 16-31: rows M=lane-16, K = k0+8..15 and k0+24..31
  int row = m0 + (lane & 15);
  int kb  = (lane < 16) ? 0 : 8;
  const __bf16* p = A + (size_t)row * lda + k0 + kb;
  FragBF f;
  f.q[0] = *(const uint4*)(p);        // 8 bf16: K = k0+kb .. +7
  f.q[1] = *(const uint4*)(p + 16);   // 8 bf16: K = k0+kb+16 .. +23
  return f.v;
}

__device__ __forceinline__ v16bf load_frag_b(const __bf16* __restrict__ Bt, int ldb,
                                             int n0, int k0, int lane) {
  // Bt is [N,K] (column-major B). lanes 0-15: col=lane, K=k0+0..7 and k0+16..23
  // lanes 16-31: col=lane-16, K=k0+8..15 and k0+24..31
  int col = n0 + (lane & 15);
  int kb  = (lane < 16) ? 0 : 8;
  const __bf16* p = Bt + (size_t)col * ldb + k0 + kb;
  FragBF f;
  f.q[0] = *(const uint4*)(p);
  f.q[1] = *(const uint4*)(p + 16);
  return f.v;
}

// ---------------- GEMM: C[M,N] = A_bf16[M,Kpad] x Bt_bf16[N,Kpad] (+bias) ---
// block = 256 threads (8 waves). wave -> 16 rows x 64 cols (4 WMMA tiles,
// shared A fragment). block tile = 128 x 64.

__global__ void gat_wmma_gemm(const __bf16* __restrict__ A,
                              const __bf16* __restrict__ Bt,
                              float* __restrict__ C,
                              const float* __restrict__ bias,
                              int M, int K, int lda, int ldb, int ldc) {
  const int lane = threadIdx.x & 31;
  const int wave = threadIdx.x >> 5;
  const int m0 = (blockIdx.x * 8 + wave) * 16;
  const int n0 = blockIdx.y * 64;
  if (m0 >= M) return;                       // wave-uniform exit

  v8f acc[4];
  acc[0] = v8f{}; acc[1] = v8f{}; acc[2] = v8f{}; acc[3] = v8f{};

  for (int k0 = 0; k0 < K; k0 += 32) {
    v16bf a  = load_frag_a(A, lda, m0, k0, lane);
    v16bf b0 = load_frag_b(Bt, ldb, n0 +  0, k0, lane);
    v16bf b1 = load_frag_b(Bt, ldb, n0 + 16, k0, lane);
    v16bf b2 = load_frag_b(Bt, ldb, n0 + 32, k0, lane);
    v16bf b3 = load_frag_b(Bt, ldb, n0 + 48, k0, lane);
    acc[0] = __builtin_amdgcn_wmma_f32_16x16x32_bf16(false, a, false, b0, (short)0, acc[0], false, false);
    acc[1] = __builtin_amdgcn_wmma_f32_16x16x32_bf16(false, a, false, b1, (short)0, acc[1], false, false);
    acc[2] = __builtin_amdgcn_wmma_f32_16x16x32_bf16(false, a, false, b2, (short)0, acc[2], false, false);
    acc[3] = __builtin_amdgcn_wmma_f32_16x16x32_bf16(false, a, false, b3, (short)0, acc[3], false, false);
  }

  const int rbase = m0 + ((lane < 16) ? 0 : 8);
  const int cl    = lane & 15;
#pragma unroll
  for (int t = 0; t < 4; ++t) {
    int col = n0 + t * 16 + cl;
    float bv = bias ? bias[col] : 0.0f;
#pragma unroll
    for (int r = 0; r < 8; ++r)
      C[(size_t)(rbase + r) * ldc + col] = acc[t][r] + bv;
  }
}

// ---------------- conversion / transpose helpers ---------------------------

__global__ void gat_cvt_bf16_pad(const float* __restrict__ src, __bf16* __restrict__ dst,
                                 int rows, int cols, int colsp) {
  int idx = blockIdx.x * blockDim.x + threadIdx.x;
  if (idx >= rows * colsp) return;
  int r = idx / colsp, c = idx - r * colsp;
  dst[idx] = (c < cols) ? (__bf16)src[(size_t)r * cols + c] : (__bf16)0.0f;
}

// src is [K,N] row-major; dst is [N,KP] bf16 (zero pad K->KP)
__global__ void gat_transpose_bf16(const float* __restrict__ src, __bf16* __restrict__ dst,
                                   int K, int N, int KP) {
  int idx = blockIdx.x * blockDim.x + threadIdx.x;
  if (idx >= N * KP) return;
  int n = idx / KP, k = idx - n * KP;
  dst[idx] = (k < K) ? (__bf16)src[(size_t)k * N + n] : (__bf16)0.0f;
}

__global__ void gat_fill_u32(unsigned* __restrict__ p, unsigned v, int n) {
  int i = blockIdx.x * blockDim.x + threadIdx.x;
  if (i < n) p[i] = v;
}

// ---------------- edge phase ----------------------------------------------

__device__ __forceinline__ void edge_pair(const int* __restrict__ ei, int E, int eid,
                                          int& src, int& dst) {
  if (eid < E) { src = ei[eid]; dst = ei[E + eid]; }
  else         { src = dst = eid - E; }           // self loops appended
}

// pass 1: e[eid,h] = sum_c leakyrelu(xl[src,h,c]+xr[dst,h,c]) * att[h,c]
// and segment max into monotonic-uint keys.
__global__ void gat_edge_scores(const float* __restrict__ xl, const float* __restrict__ xr,
                                const float* __restrict__ att, const int* __restrict__ ei,
                                int E, int Etot,
                                float* __restrict__ ebuf, unsigned* __restrict__ emaxk) {
  int eid  = blockIdx.x;
  int h    = threadIdx.x >> 5;
  int lane = threadIdx.x & 31;
  if (eid >= Etot) return;
  int src, dst; edge_pair(ei, E, eid, src, dst);
  const float* pl = xl + (size_t)src * HC + h * CCH;
  const float* pr = xr + (size_t)dst * HC + h * CCH;
  const float* pa = att + h * CCH;
  float s = 0.0f;
#pragma unroll 2
  for (int c = lane; c < CCH; c += 32) {
    float v = pl[c] + pr[c];
    v = (v > 0.0f) ? v : NEG_SLOPE * v;
    s += v * pa[c];
  }
#pragma unroll
  for (int off = 16; off; off >>= 1) s += __shfl_xor(s, off, 32);
  if (lane == 0) {
    ebuf[eid * NHEAD + h] = s;
    unsigned b = __float_as_uint(s);
    unsigned key = (b & 0x80000000u) ? ~b : (b | 0x80000000u); // order-preserving
    atomicMax(&emaxk[dst * NHEAD + h], key);
  }
}

// pass 2: ee = exp(e - emax[dst]); denom[dst] += ee; ebuf <- ee
__global__ void gat_edge_softmax(const int* __restrict__ ei, int E, int Etot,
                                 const unsigned* __restrict__ emaxk,
                                 float* __restrict__ ebuf, float* __restrict__ denom) {
  int idx = blockIdx.x * blockDim.x + threadIdx.x;
  if (idx >= Etot * NHEAD) return;
  int eid = idx >> 3, h = idx & 7;
  int src, dst; edge_pair(ei, E, eid, src, dst); (void)src;
  unsigned k = emaxk[dst * NHEAD + h];
  float mx = (k & 0x80000000u) ? __uint_as_float(k & 0x7FFFFFFFu) : __uint_as_float(~k);
  float ee = __expf(ebuf[idx] - mx);
  ebuf[idx] = ee;
  atomicAdd(&denom[dst * NHEAD + h], ee);
}

// pass 3: acc[dst,h,:] += (ee/denom) * xl[src,h,:]
__global__ void gat_edge_scatter(const float* __restrict__ xl, const int* __restrict__ ei,
                                 int E, int Etot, const float* __restrict__ ebuf,
                                 const float* __restrict__ denom, float* __restrict__ acc) {
  int eid  = blockIdx.x;
  int h    = threadIdx.x >> 5;
  int lane = threadIdx.x & 31;
  if (eid >= Etot) return;
  int src, dst; edge_pair(ei, E, eid, src, dst);
  float alpha = ebuf[eid * NHEAD + h] / (denom[dst * NHEAD + h] + 1e-16f);
  const float* pl = xl  + (size_t)src * HC + h * CCH;
  float*       po = acc + (size_t)dst * HC + h * CCH;
#pragma unroll 2
  for (int c = lane; c < CCH; c += 32)
    atomicAdd(&po[c], alpha * pl[c]);
}

// head mean + bias -> new node features
__global__ void gat_head_mean(const float* __restrict__ acc, const float* __restrict__ bias,
                              float* __restrict__ hout, int n) {
  int idx = blockIdx.x * blockDim.x + threadIdx.x;
  if (idx >= n * CCH) return;
  int node = idx >> 8, c = idx & 255;
  const float* p = acc + (size_t)node * HC + c;
  float s = 0.0f;
#pragma unroll
  for (int h = 0; h < NHEAD; ++h) s += p[h * CCH];
  hout[idx] = s * (1.0f / NHEAD) + bias[c];
}

// ---------------- pooling + head ------------------------------------------

__global__ void gat_pool(const float* __restrict__ h, const int* __restrict__ batch,
                         float* __restrict__ pooled, float* __restrict__ cnt, int n) {
  int node = blockIdx.x;
  int c    = threadIdx.x;
  if (node >= n) return;
  int g = batch[node];
  atomicAdd(&pooled[g * CCH + c], h[(size_t)node * CCH + c]);
  if (c == 0) atomicAdd(&cnt[g], 1.0f);
}

__global__ void gat_head_out(const float* __restrict__ pooled, const float* __restrict__ cnt,
                             const float* __restrict__ hw, const float* __restrict__ hb,
                             float* __restrict__ out) {
  int g = blockIdx.x;
  int lane = threadIdx.x;
  float inv = 1.0f / fmaxf(cnt[g], 1.0f);
  float s = 0.0f;
  for (int c = lane; c < CCH; c += 32) s += pooled[g * CCH + c] * inv * hw[c];
#pragma unroll
  for (int off = 16; off; off >>= 1) s += __shfl_xor(s, off, 32);
  if (lane == 0) out[g] = s + hb[0];
}

// ---------------------------------------------------------------------------

static inline char* carve(char*& p, size_t bytes) {
  char* r = p;
  p += (bytes + 255) & ~(size_t)255;
  return r;
}

extern "C" void kernel_launch(void* const* d_in, const int* in_sizes, int n_in,
                              void* d_out, int out_size, void* d_ws, size_t ws_size,
                              hipStream_t stream) {
  const float* x        = (const float*)d_in[0];
  const int*   ei       = (const int*)  d_in[1];
  const int*   batch    = (const int*)  d_in[2];
  const float* dense_w  = (const float*)d_in[3];
  const float* dense_b  = (const float*)d_in[4];
  const float* conv_wl  = (const float*)d_in[5];
  const float* conv_wr  = (const float*)d_in[6];
  const float* conv_att = (const float*)d_in[7];
  const float* conv_b   = (const float*)d_in[8];
  const float* head_w   = (const float*)d_in[9];
  const float* head_b   = (const float*)d_in[10];
  float* out = (float*)d_out;

  const int nNodes = in_sizes[0] / FEAT;       // 10000
  const int E      = in_sizes[1] / 2;          // 160000
  const int Etot   = E + nNodes;               // + self loops

  // ---- workspace layout ----
  char* p = (char*)d_ws;
  float*   h_feat = (float*)  carve(p, (size_t)nNodes * CCH * 4);
  __bf16*  hbf    = (__bf16*) carve(p, (size_t)nNodes * CCH * 2);
  __bf16*  xbf    = (__bf16*) carve(p, (size_t)nNodes * KPAD * 2);
  __bf16*  wdT    = (__bf16*) carve(p, (size_t)CCH * KPAD * 2);
  __bf16*  wlT    = (__bf16*) carve(p, (size_t)NLAYER * HC * CCH * 2);
  __bf16*  wrT    = (__bf16*) carve(p, (size_t)NLAYER * HC * CCH * 2);
  float*   xl     = (float*)  carve(p, (size_t)nNodes * HC * 4);
  float*   xr     = (float*)  carve(p, (size_t)nNodes * HC * 4);   // also scatter acc
  float*   ebuf   = (float*)  carve(p, (size_t)Etot * NHEAD * 4);
  unsigned* emaxk = (unsigned*)carve(p, (size_t)nNodes * NHEAD * 4);
  float*   denom  = (float*)  carve(p, (size_t)nNodes * NHEAD * 4);
  float*   pooled = (float*)  carve(p, (size_t)NGRAPH * CCH * 4);
  float*   cnt    = (float*)  carve(p, (size_t)NGRAPH * 4);
  float*   acc    = xr;  // lifetimes of xr and acc are disjoint within a layer
  (void)ws_size; (void)n_in; (void)out_size;

  const int T = 256;
  auto blocks = [](long long n, int t) { return (unsigned)((n + t - 1) / t); };

  // ---- weight prep ----
  gat_transpose_bf16<<<blocks((long long)CCH * KPAD, T), T, 0, stream>>>(dense_w, wdT, FEAT, CCH, KPAD);
  gat_cvt_bf16_pad  <<<blocks((long long)nNodes * KPAD, T), T, 0, stream>>>(x, xbf, nNodes, FEAT, KPAD);
  for (int l = 0; l < NLAYER; ++l) {
    gat_transpose_bf16<<<blocks((long long)HC * CCH, T), T, 0, stream>>>(
        conv_wl + (size_t)l * CCH * HC, wlT + (size_t)l * HC * CCH, CCH, HC, CCH);
    gat_transpose_bf16<<<blocks((long long)HC * CCH, T), T, 0, stream>>>(
        conv_wr + (size_t)l * CCH * HC, wrT + (size_t)l * HC * CCH, CCH, HC, CCH);
  }

  // ---- dense layer: h = x @ dense_w + dense_b ----
  {
    dim3 g((nNodes + 127) / 128, CCH / 64);
    gat_wmma_gemm<<<g, T, 0, stream>>>(xbf, wdT, h_feat, dense_b,
                                       nNodes, KPAD, KPAD, KPAD, CCH);
  }

  // ---- GATv2 layers ----
  for (int l = 0; l < NLAYER; ++l) {
    gat_cvt_bf16_pad<<<blocks((long long)nNodes * CCH, T), T, 0, stream>>>(
        h_feat, hbf, nNodes, CCH, CCH);

    dim3 g((nNodes + 127) / 128, HC / 64);
    gat_wmma_gemm<<<g, T, 0, stream>>>(hbf, wlT + (size_t)l * HC * CCH, xl, nullptr,
                                       nNodes, CCH, CCH, CCH, HC);
    gat_wmma_gemm<<<g, T, 0, stream>>>(hbf, wrT + (size_t)l * HC * CCH, xr, nullptr,
                                       nNodes, CCH, CCH, CCH, HC);

    gat_fill_u32<<<blocks((long long)nNodes * NHEAD, T), T, 0, stream>>>(emaxk, 0u, nNodes * NHEAD);
    gat_fill_u32<<<blocks((long long)nNodes * NHEAD, T), T, 0, stream>>>((unsigned*)denom, 0u, nNodes * NHEAD);

    gat_edge_scores<<<Etot, T, 0, stream>>>(xl, xr, conv_att + (size_t)l * NHEAD * CCH,
                                            ei, E, Etot, ebuf, emaxk);
    gat_edge_softmax<<<blocks((long long)Etot * NHEAD, T), T, 0, stream>>>(
        ei, E, Etot, emaxk, ebuf, denom);

    gat_fill_u32<<<blocks((long long)nNodes * HC, T), T, 0, stream>>>((unsigned*)acc, 0u, nNodes * HC);
    gat_edge_scatter<<<Etot, T, 0, stream>>>(xl, ei, E, Etot, ebuf, denom, acc);

    gat_head_mean<<<blocks((long long)nNodes * CCH, T), T, 0, stream>>>(
        acc, conv_b + (size_t)l * CCH, h_feat, nNodes);
  }

  // ---- global mean pool + linear head ----
  gat_fill_u32<<<blocks(NGRAPH * CCH + NGRAPH, T), T, 0, stream>>>((unsigned*)pooled, 0u, NGRAPH * CCH + NGRAPH);
  gat_pool<<<nNodes, CCH, 0, stream>>>(h_feat, batch, pooled, cnt, nNodes);
  gat_head_out<<<NGRAPH, 32, 0, stream>>>(pooled, cnt, head_w, head_b, out);
}
